// HeteroAttentionAggregationLayer_45174466019349
// MI455X (gfx1250) — matrix operand order repaired and checked by reading, not compiled, not run on hardware
//
#include <hip/hip_runtime.h>
#include <hip/hip_bf16.h>

// ---------------------------------------------------------------------------
// HeteroAttentionAggregationLayer for MI455X (gfx1250, wave32, WMMA).
// GEMMs: bf16 WMMA (v_wmma_f32_16x16x32_bf16, fp32 accumulate), LDS-staged A
// tiles (padded stride, conflict-free ds_load_b128), weights pre-swizzled to
// per-lane fragment order (coalesced 32B loads). All A fragments for a block
// tile are prefetched from LDS before the WMMA chain so the 8 matrix ops issue
// back-to-back (staged dscnt waits, XDL/VALU co-execution). Stores are guarded
// with wave-uniform scalar branches; EXEC stays all-ones around WMMA.
// Sparse edge phase: wave-per-edge float4 gathers + lane-group reductions,
// ordered-uint atomic max for segment-max, float atomics for sums/scatter.
// ---------------------------------------------------------------------------

#define N_ITEM 100000
#define N_USER 50000
#define DIM    128
#define NHEAD  8
#define E1N    500000
#define E2N    250000
#define ETOT   (E1N + E2N)
#define LDS_STRIDE 136   // bf16 elements per staged row (128 + 8 pad -> bank shift 4)

typedef __attribute__((ext_vector_type(16))) __bf16 v16bf;
typedef __attribute__((ext_vector_type(8)))  float  v8f;

union ABFrag { v16bf v; unsigned u[8]; uint4 q[2]; };

static __device__ __forceinline__ __bf16 f2bf(float f) {
  unsigned u = __builtin_bit_cast(unsigned, f);
  unsigned r = u + 0x7FFFu + ((u >> 16) & 1u);          // round-to-nearest-even
  unsigned short h = (unsigned short)(r >> 16);
  return __builtin_bit_cast(__bf16, h);
}

static __device__ __forceinline__ unsigned fenc(float f) {
  unsigned u = __builtin_bit_cast(unsigned, f);
  return (u & 0x80000000u) ? ~u : (u | 0x80000000u);    // order-preserving map
}
static __device__ __forceinline__ float fdec(unsigned e) {
  unsigned u = (e & 0x80000000u) ? (e ^ 0x80000000u) : ~e;
  return __builtin_bit_cast(float, u);
}

static __device__ __forceinline__ float sigm(float x) {
  return 1.0f / (1.0f + __expf(-x));
}

// ------------------------------ utility ------------------------------------

__global__ void zero_u32(unsigned* p, long long n) {
  long long i = (long long)blockIdx.x * blockDim.x + threadIdx.x;
  long long stride = (long long)gridDim.x * blockDim.x;
  for (; i < n; i += stride) p[i] = 0u;
}

// ------------------------------ batch norm ---------------------------------

// 128 threads/block: thread j owns column j over a row chunk (coalesced).
__global__ void bn_partial(const float* __restrict__ x, int n,
                           float* __restrict__ sum, float* __restrict__ sumsq) {
  int col = threadIdx.x;
  int rows_per_blk = (n + gridDim.x - 1) / gridDim.x;
  int r0 = blockIdx.x * rows_per_blk;
  int r1 = r0 + rows_per_blk; if (r1 > n) r1 = n;
  float s = 0.f, q = 0.f;
  for (int r = r0; r < r1; ++r) {
    float v = x[(size_t)r * DIM + col];
    s += v; q += v * v;
  }
  atomicAdd(&sum[col], s);
  atomicAdd(&sumsq[col], q);
}

__global__ void bn_finalize(const float* __restrict__ sum, const float* __restrict__ sumsq,
                            const float* __restrict__ g, const float* __restrict__ b,
                            float n, float* __restrict__ scale, float* __restrict__ shift) {
  int j = threadIdx.x;
  float mu  = sum[j] / n;
  float var = sumsq[j] / n - mu * mu;
  float rs  = rsqrtf(var + 1e-5f) * g[j];
  scale[j] = rs;
  shift[j] = b[j] - mu * rs;
}

__global__ void norm_to_bf16(const float* __restrict__ x, const float* __restrict__ scale,
                             const float* __restrict__ shift, __bf16* __restrict__ out,
                             long long total) {
  long long i = (long long)blockIdx.x * blockDim.x + threadIdx.x;
  if (i >= total) return;
  int j = (int)(i & (DIM - 1));
  out[i] = f2bf(x[i] * scale[j] + shift[j]);
}

// Weight [j][k] -> bf16 B-fragments in per-lane order:
//   out[(((w*4 + kc)*32 + lane)*16 + e] = W[j=w*16+(lane&15)][k=kc*32+(lane>>4)*16+e]
// so a wave's B fragment for (n-strip w, k-chunk kc) is one contiguous 32B/lane load.
__global__ void wt_to_frag(const float* __restrict__ W, __bf16* __restrict__ out) {
  int i = blockIdx.x * blockDim.x + threadIdx.x;   // 16384 threads
  int e    = i & 15;
  int lane = (i >> 4) & 31;
  int kc   = (i >> 9) & 3;
  int w    = (i >> 11) & 7;
  int k = kc * 32 + (lane >> 4) * 16 + e;
  int j = w * 16 + (lane & 15);
  out[i] = f2bf(W[j * DIM + k]);
}

__global__ void f32_to_bf16(const float* __restrict__ x, __bf16* __restrict__ out,
                            long long n) {
  long long i = (long long)blockIdx.x * blockDim.x + threadIdx.x;
  if (i < n) out[i] = f2bf(x[i]);
}

// ------------------------------ WMMA GEMM ----------------------------------
// C[M x 128] = A[M x 128] (bf16) * W  (+ optional A2*W2, + bias, ReLU).
// Block = 256 threads = 8 waves; block tile = 32 rows x 128 cols; wave owns a
// 16-wide N strip and both 16-row subtiles. A staged via LDS (padded stride).

static __device__ __forceinline__ void stage_a(const __bf16* __restrict__ A, int M,
                                               int row0, int tid, __bf16* smem) {
  int Mc = M - 1;
#pragma unroll
  for (int it = 0; it < 2; ++it) {                // 512 (row,chunk) pairs / 256 thr
    int idx = tid + it * 256;
    int r = idx >> 4, ch = idx & 15;
    int grow = row0 + r; grow = grow > Mc ? Mc : grow;   // branchless clamp
    *(uint4*)(smem + r * LDS_STRIDE + ch * 8) =
        *(const uint4*)(A + (size_t)grow * DIM + ch * 8);
  }
}

// Prefetch all 8 A fragments (2 subtiles x 4 k-chunks) from LDS, then run the
// 8 WMMAs back-to-back. Garbage rows (beyond M) are computed but never stored.
static __device__ __forceinline__ void wmma_tile(const __bf16* __restrict__ smem,
                                                 const v16bf bf[4], v8f acc[2],
                                                 int l15, int half) {
  ABFrag a[8];
#pragma unroll
  for (int s = 0; s < 2; ++s) {
#pragma unroll
    for (int kc = 0; kc < 4; ++kc) {
      const __bf16* base = smem + (s * 16 + l15) * LDS_STRIDE + kc * 32 + half * 16;
      a[s * 4 + kc].q[0] = *(const uint4*)(base);        // K offsets 0..7
      a[s * 4 + kc].q[1] = *(const uint4*)(base + 8);    // K offsets 8..15
    }
  }
#pragma unroll
  for (int s = 0; s < 2; ++s)
#pragma unroll
    for (int kc = 0; kc < 4; ++kc)
      acc[s] = __builtin_amdgcn_wmma_f32_16x16x32_bf16(false, a[s * 4 + kc].v, false,
                                                       bf[kc], (short)0, acc[s],
                                                       false, false);
}

__global__ void __launch_bounds__(256)
gemm_bf16_wmma(const __bf16* __restrict__ A,  const __bf16* __restrict__ Bf,
               const __bf16* __restrict__ A2, const __bf16* __restrict__ B2f,
               const float* __restrict__ bias, float* __restrict__ C,
               int M, int do_relu) {
  __shared__ __bf16 smem[32 * LDS_STRIDE];
  int tid  = threadIdx.x;
  int lane = tid & 31;
  int wave = tid >> 5;                 // 0..7 -> N strip
  int half = lane >> 4;
  int l15  = lane & 15;
  int row0 = blockIdx.x * 32;

  // B fragments for this wave's strip: contiguous 32B per lane, coalesced.
  v16bf bf[4];
#pragma unroll
  for (int kc = 0; kc < 4; ++kc)
    bf[kc] = *(const v16bf*)(Bf + ((((wave * 4 + kc) * 32) + lane) << 4));

  v8f acc[2] = {{}, {}};

  stage_a(A, M, row0, tid, smem);
  __syncthreads();
  wmma_tile(smem, bf, acc, l15, half);

  if (A2 != nullptr) {                 // uniform branch
    __syncthreads();
    stage_a(A2, M, row0, tid, smem);
    __syncthreads();
    v16bf b2[4];
#pragma unroll
    for (int kc = 0; kc < 4; ++kc)
      b2[kc] = *(const v16bf*)(B2f + ((((wave * 4 + kc) * 32) + lane) << 4));
    wmma_tile(smem, b2, acc, l15, half);
  }

  int colw = wave * 16 + l15;
  float bv = (bias != nullptr) ? bias[colw] : 0.0f;
#pragma unroll
  for (int s = 0; s < 2; ++s) {
    int rm0 = row0 + s * 16;
    if (rm0 < M) {                     // wave-uniform full-subtile guard (M%16==0)
#pragma unroll
      for (int r = 0; r < 8; ++r) {    // C/D layout: VGPR r -> rows r / r+8
        float v = acc[s][r] + bv;
        if (do_relu) v = fmaxf(v, 0.0f);
        C[(size_t)(rm0 + r + half * 8) * DIM + colw] = v;
      }
    }
  }
}

// ------------------------------ edge phase ---------------------------------

// wave per edge; lane handles 4 contiguous dims; 4-lane reduce -> per-head score
__global__ void edge_scores(const float* __restrict__ q, const float* __restrict__ k,
                            const float* __restrict__ ae,
                            const int* __restrict__ uid, const int* __restrict__ vid,
                            const int* __restrict__ cnt, const float* __restrict__ emb,
                            float* __restrict__ sc, int E) {
  int lane = threadIdx.x & 31;
  int e = blockIdx.x * 8 + (threadIdx.x >> 5);
  if (e >= E) return;
  int u = uid[e], v = vid[e];
  float4 qv = ((const float4*)(q + (size_t)u * DIM))[lane];
  float4 kv = ((const float4*)(k + (size_t)v * DIM))[lane];
  float x0 = qv.x + kv.x, x1 = qv.y + kv.y, x2 = qv.z + kv.z, x3 = qv.w + kv.w;
  if (cnt != nullptr) {
    float4 ev = ((const float4*)(emb + (size_t)cnt[e] * DIM))[lane];
    x0 += ev.x; x1 += ev.y; x2 += ev.z; x3 += ev.w;
  }
  float4 av = ((const float4*)ae)[lane];
  float p = av.x * sigm(x0) + av.y * sigm(x1) + av.z * sigm(x2) + av.w * sigm(x3);
  p += __shfl_xor(p, 1, 32);
  p += __shfl_xor(p, 2, 32);
  if ((lane & 3) == 0) sc[(size_t)e * NHEAD + (lane >> 2)] = p;
}

__global__ void seg_max(const float* __restrict__ sc,
                        const int* __restrict__ vid1, const int* __restrict__ vid2,
                        unsigned* __restrict__ maxu) {
  long long i = (long long)blockIdx.x * blockDim.x + threadIdx.x;
  if (i >= (long long)ETOT * NHEAD) return;
  int e = (int)(i >> 3), h = (int)(i & 7);
  int dst = (e < E1N) ? vid1[e] : vid2[e - E1N];
  atomicMax(&maxu[dst * NHEAD + h], fenc(sc[i]));
}

__global__ void exp_seg_sum(float* __restrict__ sc,
                            const int* __restrict__ vid1, const int* __restrict__ vid2,
                            const unsigned* __restrict__ maxu, float* __restrict__ sumb) {
  long long i = (long long)blockIdx.x * blockDim.x + threadIdx.x;
  if (i >= (long long)ETOT * NHEAD) return;
  int e = (int)(i >> 3), h = (int)(i & 7);
  int dst = (e < E1N) ? vid1[e] : vid2[e - E1N];
  float m = fdec(maxu[dst * NHEAD + h]);
  float ex = __expf(sc[i] - m);
  sc[i] = ex;
  atomicAdd(&sumb[dst * NHEAD + h], ex);
}

// wave per edge: agg[dst] += v_src * a (per head), 4 float atomics per lane
__global__ void scatter_agg(const float* __restrict__ ex, const float* __restrict__ sumb,
                            const float* __restrict__ v1, const float* __restrict__ v2,
                            const int* __restrict__ uid1, const int* __restrict__ vid1,
                            const int* __restrict__ uid2, const int* __restrict__ vid2,
                            float* __restrict__ agg) {
  int lane = threadIdx.x & 31;
  int e = blockIdx.x * 8 + (threadIdx.x >> 5);
  if (e >= ETOT) return;
  int src, dst; const float* val;
  if (e < E1N) { src = uid1[e];       dst = vid1[e];       val = v1 + (size_t)src * DIM; }
  else         { int t = e - E1N; src = uid2[t]; dst = vid2[t]; val = v2 + (size_t)src * DIM; }
  int h = lane >> 2;
  float a = ex[(size_t)e * NHEAD + h] / sumb[dst * NHEAD + h];
  float4 vv = ((const float4*)val)[lane];
  float* out = agg + (size_t)dst * DIM + lane * 4;
  atomicAdd(out + 0, vv.x * a);
  atomicAdd(out + 1, vv.y * a);
  atomicAdd(out + 2, vv.z * a);
  atomicAdd(out + 3, vv.w * a);
}

// ------------------------------ launcher -----------------------------------

extern "C" void kernel_launch(void* const* d_in, const int* in_sizes, int n_in,
                              void* d_out, int out_size, void* d_ws, size_t ws_size,
                              hipStream_t stream) {
  const float* ft_item   = (const float*)d_in[0];
  const float* ft_user   = (const float*)d_in[1];
  const float* bn_g_item = (const float*)d_in[2];
  const float* bn_b_item = (const float*)d_in[3];
  const float* bn_g_user = (const float*)d_in[4];
  const float* bn_b_user = (const float*)d_in[5];
  const float* Wq1 = (const float*)d_in[6];
  const float* bq1 = (const float*)d_in[7];
  const float* Wk1 = (const float*)d_in[8];
  const float* Wv1 = (const float*)d_in[9];
  const float* ae1 = (const float*)d_in[10];
  const float* emb_cnt1 = (const float*)d_in[11];
  const float* Wq2 = (const float*)d_in[12];
  const float* bq2 = (const float*)d_in[13];
  const float* Wk2 = (const float*)d_in[14];
  const float* Wv2 = (const float*)d_in[15];
  const float* ae2 = (const float*)d_in[16];
  const float* W_agg  = (const float*)d_in[17];
  const float* b_agg  = (const float*)d_in[18];
  const float* W_self = (const float*)d_in[19];
  const int* uid1 = (const int*)d_in[20];
  const int* vid1 = (const int*)d_in[21];
  const int* cnt1 = (const int*)d_in[22];
  const int* uid2 = (const int*)d_in[23];
  const int* vid2 = (const int*)d_in[24];
  float* out = (float*)d_out;

  // ---- workspace layout (512B aligned slabs) ----
  char* ws = (char*)d_ws;
  auto alloc = [&](size_t bytes) -> char* {
    char* p = ws; ws += (bytes + 511) & ~((size_t)511); return p;
  };
  float* sum_i  = (float*)alloc(DIM * 4 * sizeof(float));   // sum_i, sq_i, sum_u, sq_u
  float* sq_i   = sum_i + DIM;
  float* sum_u  = sum_i + 2 * DIM;
  float* sq_u   = sum_i + 3 * DIM;
  float* scale_i = (float*)alloc(DIM * 4 * sizeof(float));  // scale/shift item+user
  float* shift_i = scale_i + DIM;
  float* scale_u = scale_i + 2 * DIM;
  float* shift_u = scale_i + 3 * DIM;

  __bf16* xi_bf = (__bf16*)alloc((size_t)N_ITEM * DIM * 2);
  __bf16* xu_bf = (__bf16*)alloc((size_t)N_USER * DIM * 2);
  __bf16* wq1t  = (__bf16*)alloc(DIM * DIM * 2);
  __bf16* wk1t  = (__bf16*)alloc(DIM * DIM * 2);
  __bf16* wv1t  = (__bf16*)alloc(DIM * DIM * 2);
  __bf16* wq2t  = (__bf16*)alloc(DIM * DIM * 2);
  __bf16* wk2t  = (__bf16*)alloc(DIM * DIM * 2);
  __bf16* wv2t  = (__bf16*)alloc(DIM * DIM * 2);
  __bf16* waggt = (__bf16*)alloc(DIM * DIM * 2);
  __bf16* wselft= (__bf16*)alloc(DIM * DIM * 2);
  float* q1 = (float*)alloc((size_t)N_ITEM * DIM * 4);
  float* k1 = (float*)alloc((size_t)N_ITEM * DIM * 4);
  float* v1 = (float*)alloc((size_t)N_ITEM * DIM * 4);
  float* k2 = (float*)alloc((size_t)N_ITEM * DIM * 4);
  float* q2 = (float*)alloc((size_t)N_USER * DIM * 4);
  float* v2 = (float*)alloc((size_t)N_USER * DIM * 4);
  float* sc = (float*)alloc((size_t)ETOT * NHEAD * 4);
  unsigned* maxu = (unsigned*)alloc((size_t)N_ITEM * NHEAD * 4);
  float* sumb    = (float*)alloc((size_t)N_ITEM * NHEAD * 4);   // contiguous w/ maxu
  float* agg     = (float*)alloc((size_t)N_ITEM * DIM * 4);     // contiguous w/ sumb
  __bf16* agg_bf = (__bf16*)alloc((size_t)N_ITEM * DIM * 2);
  (void)ws_size; (void)n_in; (void)in_sizes; (void)out_size;

  // ---- 0: zero stat accumulators + max/sum/agg region ----
  zero_u32<<<8, 256, 0, stream>>>((unsigned*)sum_i, 4 * DIM);
  {
    long long nwords = (long long)N_ITEM * NHEAD * 2 + (long long)N_ITEM * DIM;
    zero_u32<<<2048, 256, 0, stream>>>((unsigned*)maxu, nwords);
  }

  // ---- 1: batch-norm statistics ----
  bn_partial<<<512, DIM, 0, stream>>>(ft_item, N_ITEM, sum_i, sq_i);
  bn_partial<<<512, DIM, 0, stream>>>(ft_user, N_USER, sum_u, sq_u);
  bn_finalize<<<1, DIM, 0, stream>>>(sum_i, sq_i, bn_g_item, bn_b_item,
                                     (float)N_ITEM, scale_i, shift_i);
  bn_finalize<<<1, DIM, 0, stream>>>(sum_u, sq_u, bn_g_user, bn_b_user,
                                     (float)N_USER, scale_u, shift_u);

  // ---- 2: normalize + convert activations to bf16 ----
  {
    long long ti = (long long)N_ITEM * DIM, tu = (long long)N_USER * DIM;
    norm_to_bf16<<<(int)((ti + 255) / 256), 256, 0, stream>>>(ft_item, scale_i, shift_i, xi_bf, ti);
    norm_to_bf16<<<(int)((tu + 255) / 256), 256, 0, stream>>>(ft_user, scale_u, shift_u, xu_bf, tu);
  }

  // ---- 3: weights -> bf16 per-lane B-fragment order ----
  wt_to_frag<<<64, 256, 0, stream>>>(Wq1, wq1t);
  wt_to_frag<<<64, 256, 0, stream>>>(Wk1, wk1t);
  wt_to_frag<<<64, 256, 0, stream>>>(Wv1, wv1t);
  wt_to_frag<<<64, 256, 0, stream>>>(Wq2, wq2t);
  wt_to_frag<<<64, 256, 0, stream>>>(Wk2, wk2t);
  wt_to_frag<<<64, 256, 0, stream>>>(Wv2, wv2t);
  wt_to_frag<<<64, 256, 0, stream>>>(W_agg, waggt);
  wt_to_frag<<<64, 256, 0, stream>>>(W_self, wselft);

  // ---- 4: projection GEMMs (bf16 WMMA, fp32 accumulate) ----
  int gbI = (N_ITEM + 31) / 32, gbU = (N_USER + 31) / 32;
  gemm_bf16_wmma<<<gbI, 256, 0, stream>>>(xi_bf, wq1t, nullptr, nullptr, bq1,     q1, N_ITEM, 0);
  gemm_bf16_wmma<<<gbI, 256, 0, stream>>>(xi_bf, wk1t, nullptr, nullptr, nullptr, k1, N_ITEM, 0);
  gemm_bf16_wmma<<<gbI, 256, 0, stream>>>(xi_bf, wv1t, nullptr, nullptr, nullptr, v1, N_ITEM, 0);
  gemm_bf16_wmma<<<gbI, 256, 0, stream>>>(xi_bf, wk2t, nullptr, nullptr, nullptr, k2, N_ITEM, 0);
  gemm_bf16_wmma<<<gbU, 256, 0, stream>>>(xu_bf, wq2t, nullptr, nullptr, bq2,     q2, N_USER, 0);
  gemm_bf16_wmma<<<gbU, 256, 0, stream>>>(xu_bf, wv2t, nullptr, nullptr, nullptr, v2, N_USER, 0);

  // ---- 5: per-edge gated per-head scores ----
  edge_scores<<<(E1N + 7) / 8, 256, 0, stream>>>(q1, k1, ae1, uid1, vid1, cnt1, emb_cnt1,
                                                 sc, E1N);
  edge_scores<<<(E2N + 7) / 8, 256, 0, stream>>>(q2, k2, ae2, uid2, vid2, nullptr, nullptr,
                                                 sc + (size_t)E1N * NHEAD, E2N);

  // ---- 6: edge softmax over incoming edges per dst, per head ----
  {
    long long n = (long long)ETOT * NHEAD;
    int gb = (int)((n + 255) / 256);
    seg_max<<<gb, 256, 0, stream>>>(sc, vid1, vid2, maxu);
    exp_seg_sum<<<gb, 256, 0, stream>>>(sc, vid1, vid2, maxu, sumb);
  }

  // ---- 7: weighted scatter-sum of source values ----
  scatter_agg<<<(ETOT + 7) / 8, 256, 0, stream>>>(sc, sumb, v1, v2,
                                                  uid1, vid1, uid2, vid2, agg);

  // ---- 8: final fused dual-GEMM + bias + ReLU -> d_out ----
  f32_to_bf16<<<(int)(((long long)N_ITEM * DIM + 255) / 256), 256, 0, stream>>>(
      agg, agg_bf, (long long)N_ITEM * DIM);
  gemm_bf16_wmma<<<gbI, 256, 0, stream>>>(agg_bf, waggt, xi_bf, wselft, b_agg,
                                          out, N_ITEM, 1);
}
